// HDRDN_Mamba_8211977470170
// MI455X (gfx1250) — matrix operand order repaired and checked by reading, not compiled, not run on hardware
//
#include <hip/hip_runtime.h>
#include <hip/hip_bf16.h>

// ---------------------------------------------------------------------------
// HDRDN-Mamba forward for gfx1250 (MI455X). All GEMM-shaped work (3x3 convs as
// implicit GEMM, attention projections, mamba projections) runs through
// v_wmma_f32_16x16x32_f16, fp32 accumulate. Each GEMM wave computes 64x16
// outputs (4 M-tiles sharing one B fragment); each conv wave 32x16. Per K-step
// all fragments are loaded first (b128 clause), then converted, then all
// WMMAs issue back-to-back -- minimizing WMMA->VALU hazard NOPs.
// ---------------------------------------------------------------------------

typedef __attribute__((ext_vector_type(16))) _Float16 v16h;
typedef __attribute__((ext_vector_type(8)))  float    v8f;

union AFrag { v16h v; _Float16 e[16]; };
union CFrag { v8f  v; float     e[8]; };

#define EMB 60
#define DIN 120
#define NST 16
#define MT_G 4   // GEMM M-tiles per wave
#define MT_C 2   // conv M-tiles per wave

__device__ __forceinline__ float sigf_(float x) { return 1.f / (1.f + __expf(-x)); }
__device__ __forceinline__ int   imin_(int a, int b) { return a < b ? a : b; }
__device__ __forceinline__ int   imax_(int a, int b) { return a > b ? a : b; }

__device__ __forceinline__ void cvt4_(AFrag& f, int base, float4 v)
{
    f.e[base + 0] = (_Float16)v.x; f.e[base + 1] = (_Float16)v.y;
    f.e[base + 2] = (_Float16)v.z; f.e[base + 3] = (_Float16)v.w;
}

// ======================= generic WMMA GEMM (contiguous K) ==================
// Out[m*sCm + n*sCn] = sum_k A[m*sAm + k] * W[n*K + k] (+bias[n]) (+Res[idx])
// REQUIRES: K % 4 == 0, sAm % 4 == 0, A/W 16B-aligned (all call sites comply).
__global__ void __launch_bounds__(32)
k_gemm_wmma(const float* __restrict__ A, const float* __restrict__ W,
            const float* __restrict__ bias, const float* __restrict__ Res,
            float* __restrict__ Out,
            int M, int N, int K, int sAm, int sCm, int sCn)
{
    const int m0 = blockIdx.x * (16 * MT_G);
    const int n0 = blockIdx.y * 16;
    const int lane = threadIdx.x;
    const int half = lane >> 4;    // 0: lanes 0-15, 1: lanes 16-31
    const int l15  = lane & 15;
    CFrag acc[MT_G] = {};
    // Rows >= M / cols >= N only feed outputs the store guard drops: clamp
    // the address, no value select needed.
    const float4* __restrict__ A4[MT_G];
#pragma unroll
    for (int t = 0; t < MT_G; ++t)
        A4[t] = (const float4*)(A + (size_t)imin_(m0 + 16 * t + l15, M - 1) * sAm);
    const float4* __restrict__ B4 =
        (const float4*)(W + (size_t)imin_(n0 + l15, N - 1) * K);

    const int Kfull = K & ~31;
    const int qA = half * 2;   // float4 offset for A e0..7 (k = 4*q .. 4*q+7)
    const int qB = half * 2;   // float4 offset for B (same k order as A)

    for (int k0 = 0; k0 < Kfull; k0 += 32) {
        const int q = k0 >> 2;
        // 1) issue ALL loads (deep b128 clause, overlaps previous WMMAs)
        float4 bvv[4];
#pragma unroll
        for (int j = 0; j < 4; ++j) bvv[j] = B4[q + qB + (j & 1) + (j >> 1) * 4];
        float4 avv[MT_G][4];
#pragma unroll
        for (int t = 0; t < MT_G; ++t) {
            avv[t][0] = A4[t][q + qA + 0];
            avv[t][1] = A4[t][q + qA + 1];
            avv[t][2] = A4[t][q + 4 + qA + 0];
            avv[t][3] = A4[t][q + 4 + qA + 1];
        }
        // 2) convert everything
        AFrag b;
        cvt4_(b, 0, bvv[0]); cvt4_(b, 4, bvv[1]);
        cvt4_(b, 8, bvv[2]); cvt4_(b, 12, bvv[3]);
        AFrag a[MT_G];
#pragma unroll
        for (int t = 0; t < MT_G; ++t) {
            cvt4_(a[t], 0, avv[t][0]); cvt4_(a[t], 4, avv[t][1]);
            cvt4_(a[t], 8, avv[t][2]); cvt4_(a[t], 12, avv[t][3]);
        }
        // 3) WMMAs back-to-back (B fragment reused; no intra-step hazards)
#pragma unroll
        for (int t = 0; t < MT_G; ++t)
            acc[t].v = __builtin_amdgcn_wmma_f32_16x16x32_f16(
                false, a[t].v, false, b.v, (short)0, acc[t].v, false, false);
    }
    if (Kfull < K) {
        // Tail: b128 loads with the float4 index clamped to the last in-row
        // vector (K%4==0 keeps this inside the row), then zero-select.
        const int q = Kfull >> 2;
        const int mq = (K >> 2) - 1;
        const int kA0 = Kfull + half * 8;        // intended k of a.e[0]
        const int kA1 = Kfull + 16 + half * 8;   // intended k of a.e[8]
        const int kB  = Kfull + half * 16;       // unused: b.e[0] has k = kA0 (same order as A)
        float4 bvv[4];
        bvv[0] = B4[imin_(q + qB + 0, mq)];
        bvv[1] = B4[imin_(q + qB + 1, mq)];
        bvv[2] = B4[imin_(q + 4 + qB + 0, mq)];
        bvv[3] = B4[imin_(q + 4 + qB + 1, mq)];
        float4 avv[MT_G][4];
#pragma unroll
        for (int t = 0; t < MT_G; ++t) {
            avv[t][0] = A4[t][imin_(q + qA + 0, mq)];
            avv[t][1] = A4[t][imin_(q + qA + 1, mq)];
            avv[t][2] = A4[t][imin_(q + 4 + qA + 0, mq)];
            avv[t][3] = A4[t][imin_(q + 4 + qA + 1, mq)];
        }
        AFrag b;
#pragma unroll
        for (int i = 0; i < 16; ++i)
            b.e[i] = (_Float16)((i < 8 ? kA0 + i : kA1 + i - 8) < K ? ((const float*)bvv)[i] : 0.f);
        AFrag a[MT_G];
#pragma unroll
        for (int t = 0; t < MT_G; ++t) {
#pragma unroll
            for (int i = 0; i < 8; ++i) {
                a[t].e[i]     = (_Float16)((kA0 + i) < K ? ((const float*)avv[t])[i]     : 0.f);
                a[t].e[i + 8] = (_Float16)((kA1 + i) < K ? ((const float*)avv[t])[8 + i] : 0.f);
            }
        }
#pragma unroll
        for (int t = 0; t < MT_G; ++t)
            acc[t].v = __builtin_amdgcn_wmma_f32_16x16x32_f16(
                false, a[t].v, false, b.v, (short)0, acc[t].v, false, false);
    }
    const int cn = n0 + l15;
    if (cn < N) {
        const float bv = bias ? bias[cn] : 0.f;
#pragma unroll
        for (int t = 0; t < MT_G; ++t) {
#pragma unroll
            for (int r = 0; r < 8; ++r) {
                int cm = m0 + 16 * t + r + half * 8;
                if (cm < M) {
                    size_t idx = (size_t)cm * sCm + (size_t)cn * sCn;
                    float v = acc[t].e[r] + bv;
                    if (Res) v += Res[idx];
                    Out[idx] = v;
                }
            }
        }
    }
}

// ======================= 3x3 conv as implicit-GEMM WMMA ====================
// Always loads from a clamped in-bounds address; zero-padding via select.
__device__ __forceinline__ float conv_fetchA(const float* __restrict__ In,
                                             int Hin, int Win, int oy, int ox,
                                             int stride, int padBeg, int kk)
{
    int ci = kk / 9, rem = kk - ci * 9, r = rem / 3, c = rem - r * 3;
    int iy = oy * stride + r - padBeg;
    int ix = ox * stride + c - padBeg;
    bool ok = (iy >= 0) & (iy < Hin) & (ix >= 0) & (ix < Win);
    int iyc = imin_(imax_(iy, 0), Hin - 1);
    int ixc = imin_(imax_(ix, 0), Win - 1);
    float v = In[(size_t)ci * Hin * Win + (size_t)iyc * Win + ixc];
    return ok ? v : 0.f;
}

// In: CHW fp32, W: [Cout,Cin,3,3], Out: CHW. act: 0 linear, 1 lrelu(slope), 2 sigmoid
__global__ void __launch_bounds__(32)
k_conv3x3_wmma(const float* __restrict__ In, const float* __restrict__ W,
               const float* __restrict__ bias, float* __restrict__ Out,
               int Cin, int Hin, int Win, int Cout, int stride, int act, float slope)
{
    const int Ho = (stride == 1) ? Hin : (Hin >> 1);
    const int Wo = (stride == 1) ? Win : (Win >> 1);
    const int M  = Ho * Wo;
    const int K  = Cin * 9;
    const int padBeg = (stride == 1) ? 1 : 0;  // JAX 'SAME': stride2 pads (0,1)
    const int m0 = blockIdx.x * (16 * MT_C);
    const int n0 = blockIdx.y * 16;
    const int lane = threadIdx.x;
    const int half = lane >> 4;
    const int l15  = lane & 15;
    CFrag acc[MT_C] = {};
    int oy[MT_C], ox[MT_C];
#pragma unroll
    for (int t = 0; t < MT_C; ++t) {
        int amc = imin_(m0 + 16 * t + l15, M - 1);   // clamped pixel (address only)
        oy[t] = amc / Wo; ox[t] = amc - oy[t] * Wo;
    }
    const float* __restrict__ Brow = W + (size_t)imin_(n0 + l15, Cout - 1) * K;

    const int Kfull = K & ~31;
    const int kA0 = half * 8;
    const int kA1 = 16 + half * 8;
    const int kB  = half * 16;

    for (int k0 = 0; k0 < Kfull; k0 += 32) {
        // 1) all loads staged through float arrays
        float fb[16], fa[MT_C][16];
#pragma unroll
        for (int i = 0; i < 16; ++i) fb[i] = Brow[k0 + (i < 8 ? kA0 + i : kA1 + i - 8)];
#pragma unroll
        for (int t = 0; t < MT_C; ++t) {
#pragma unroll
            for (int i = 0; i < 8; ++i) {
                fa[t][i]     = conv_fetchA(In, Hin, Win, oy[t], ox[t], stride, padBeg, k0 + kA0 + i);
                fa[t][i + 8] = conv_fetchA(In, Hin, Win, oy[t], ox[t], stride, padBeg, k0 + kA1 + i);
            }
        }
        // 2) convert
        AFrag b, a[MT_C];
#pragma unroll
        for (int i = 0; i < 16; ++i) b.e[i] = (_Float16)fb[i];
#pragma unroll
        for (int t = 0; t < MT_C; ++t)
#pragma unroll
            for (int i = 0; i < 16; ++i) a[t].e[i] = (_Float16)fa[t][i];
        // 3) WMMAs back-to-back (weights fragment reused)
#pragma unroll
        for (int t = 0; t < MT_C; ++t)
            acc[t].v = __builtin_amdgcn_wmma_f32_16x16x32_f16(
                false, a[t].v, false, b.v, (short)0, acc[t].v, false, false);
    }
    if (Kfull < K) {
        float fb[16], fa[MT_C][16];
#pragma unroll
        for (int i = 0; i < 16; ++i) {
            int kk = Kfull + (i < 8 ? kA0 + i : kA1 + i - 8);
            float v = Brow[imin_(kk, K - 1)];
            fb[i] = (kk < K) ? v : 0.f;
        }
#pragma unroll
        for (int t = 0; t < MT_C; ++t) {
#pragma unroll
            for (int i = 0; i < 8; ++i) {
                int k1 = Kfull + kA0 + i, k2 = Kfull + kA1 + i;
                float v1 = conv_fetchA(In, Hin, Win, oy[t], ox[t], stride, padBeg, imin_(k1, K - 1));
                float v2 = conv_fetchA(In, Hin, Win, oy[t], ox[t], stride, padBeg, imin_(k2, K - 1));
                fa[t][i]     = (k1 < K) ? v1 : 0.f;
                fa[t][i + 8] = (k2 < K) ? v2 : 0.f;
            }
        }
        AFrag b, a[MT_C];
#pragma unroll
        for (int i = 0; i < 16; ++i) b.e[i] = (_Float16)fb[i];
#pragma unroll
        for (int t = 0; t < MT_C; ++t)
#pragma unroll
            for (int i = 0; i < 16; ++i) a[t].e[i] = (_Float16)fa[t][i];
#pragma unroll
        for (int t = 0; t < MT_C; ++t)
            acc[t].v = __builtin_amdgcn_wmma_f32_16x16x32_f16(
                false, a[t].v, false, b.v, (short)0, acc[t].v, false, false);
    }
    const int cn = n0 + l15;
    if (cn < Cout) {
        const float bv = bias[cn];
#pragma unroll
        for (int t = 0; t < MT_C; ++t) {
#pragma unroll
            for (int r = 0; r < 8; ++r) {
                int cm = m0 + 16 * t + r + half * 8;
                if (cm < M) {
                    float v = acc[t].e[r] + bv;
                    if (act == 1) v = (v > 0.f) ? v : slope * v;
                    else if (act == 2) v = sigf_(v);
                    Out[(size_t)cn * M + cm] = v;
                }
            }
        }
    }
}

// ======================= CHW -> HWC transpose ==============================
__global__ void k_chw2hwc(const float* __restrict__ in, float* __restrict__ out,
                          int C, int HW)
{
    int i = blockIdx.x * blockDim.x + threadIdx.x;
    if (i >= C * HW) return;
    int c = i / HW, m = i - c * HW;
    out[(size_t)m * C + c] = in[i];
}

// ======================= window cross-attention core =======================
// Qf/Kf/Vf/Ao: row-major [H*W, 60]. Projections hoisted out (linear proj
// commutes with the window gather; zero padding projects to zero features).
// One block per 8x8 window; thread = (query, head). Online softmax, dh=15.
__global__ void __launch_bounds__(256)
k_win_attn(const float* __restrict__ Qf, const float* __restrict__ Kf,
           const float* __restrict__ Vf, float* __restrict__ Ao, int Hs, int Ws)
{
    const int nw  = Ws >> 3;
    const int win = blockIdx.x;
    const int wy = win / nw, wx = win - wy * nw;
    const int t = threadIdx.x;
    const int q = t >> 2;    // 0..63
    const int h = t & 3;     // head
    const int qy = (wy << 3) + (q >> 3);
    const int qx = (wx << 3) + (q & 7);
    const int qpix = qy * Ws + qx;

    float qr[15];
    const float* qp = Qf + (size_t)qpix * EMB + h * 15;
#pragma unroll
    for (int i = 0; i < 15; ++i) qr[i] = qp[i];

    const float scale = 0.25819888974716f;  // 1/sqrt(15)
    float m = -3.0e38f, den = 0.f;
    float acc[15];
#pragma unroll
    for (int i = 0; i < 15; ++i) acc[i] = 0.f;

    for (int k = 0; k < 256; ++k) {
        int blk = k >> 6;
        int di = blk >> 1, dj = blk & 1;
        int kk = k & 63;
        int ky = (wy << 3) + (di << 3) + (kk >> 3) - 4;
        int kx = (wx << 3) + (dj << 3) + (kk & 7) - 4;
        bool inb = (ky >= 0) & (ky < Hs) & (kx >= 0) & (kx < Ws);
        int kyc = imin_(imax_(ky, 0), Hs - 1);
        int kxc = imin_(imax_(kx, 0), Ws - 1);
        size_t base = (size_t)(kyc * Ws + kxc) * EMB + h * 15;
        const float* kp = Kf + base;
        const float* vp = Vf + base;
        float s = 0.f;
#pragma unroll
        for (int i = 0; i < 15; ++i) s += qr[i] * kp[i];
        s = inb ? s * scale : 0.f;   // padded pixel: zero-projected K -> score 0
        float nm = fmaxf(m, s);
        float f = __expf(m - nm);
        float p = __expf(s - nm);
        den = den * f + p;           // padded pixel still occupies softmax mass
        float pv = inb ? p : 0.f;    // ... but contributes zero V
#pragma unroll
        for (int i = 0; i < 15; ++i) acc[i] = acc[i] * f + pv * vp[i];
        m = nm;
    }
    const float inv = 1.f / den;
    float* op = Ao + (size_t)qpix * EMB + h * 15;
#pragma unroll
    for (int i = 0; i < 15; ++i) op[i] = acc[i] * inv;
}

// ======================= elementwise helpers ==============================
__global__ void k_add(float* __restrict__ d, const float* __restrict__ s, int n)
{ int i = blockIdx.x * blockDim.x + threadIdx.x; if (i < n) d[i] += s[i]; }

__global__ void k_mul(float* __restrict__ d, const float* __restrict__ s, int n)
{ int i = blockIdx.x * blockDim.x + threadIdx.x; if (i < n) d[i] *= s[i]; }

// d[c][y][x] += s[c][y/2][x/2]  (nearest x2 upsample-add)
__global__ void k_add_up2(float* __restrict__ d, const float* __restrict__ s,
                          int C, int H, int W)
{
    int i = blockIdx.x * blockDim.x + threadIdx.x;
    int n = C * H * W;
    if (i >= n) return;
    int c = i / (H * W); int rem = i - c * H * W;
    int y = rem / W, x = rem - y * W;
    d[i] += s[(size_t)c * (H / 2) * (W / 2) + (size_t)(y >> 1) * (W / 2) + (x >> 1)];
}

// ======================= layernorm x2 (pe_ln then mb_ln) ===================
__global__ void k_double_ln(const float* __restrict__ X,  // CHW [60][HW]
                            const float* __restrict__ w1, const float* __restrict__ b1,
                            const float* __restrict__ w2, const float* __restrict__ b2,
                            float* __restrict__ T, int HW)  // row-major [HW,60]
{
    int l = blockIdx.x * blockDim.x + threadIdx.x;
    if (l >= HW) return;
    float mu = 0.f;
    for (int c = 0; c < EMB; ++c) mu += X[(size_t)c * HW + l];
    mu *= (1.f / EMB);
    float var = 0.f;
    for (int c = 0; c < EMB; ++c) { float d = X[(size_t)c * HW + l] - mu; var += d * d; }
    var *= (1.f / EMB);
    float inv1 = rsqrtf(var + 1e-5f);
    float mu2 = 0.f;
    for (int c = 0; c < EMB; ++c) {
        float y = (X[(size_t)c * HW + l] - mu) * inv1 * w1[c] + b1[c];
        mu2 += y;
    }
    mu2 *= (1.f / EMB);
    float var2 = 0.f;
    for (int c = 0; c < EMB; ++c) {
        float y = (X[(size_t)c * HW + l] - mu) * inv1 * w1[c] + b1[c];
        float d = y - mu2; var2 += d * d;
    }
    var2 *= (1.f / EMB);
    float inv2 = rsqrtf(var2 + 1e-5f);
    for (int c = 0; c < EMB; ++c) {
        float y = (X[(size_t)c * HW + l] - mu) * inv1 * w1[c] + b1[c];
        T[(size_t)l * EMB + c] = (y - mu2) * inv2 * w2[c] + b2[c];
    }
}

// ======================= mamba pieces ======================================
// u[l,d] = silu( sum_{j=0..3} xi[l-3+j,d]*w[d,j] + b[d] ), xi = xz[:, :120]
__global__ void k_conv1d_silu(const float* __restrict__ xz, const float* __restrict__ w,
                              const float* __restrict__ b, float* __restrict__ u, int L)
{
    int idx = blockIdx.x * blockDim.x + threadIdx.x;
    if (idx >= L * DIN) return;
    int l = idx / DIN, d = idx - l * DIN;
    float s = b[d];
#pragma unroll
    for (int j = 0; j < 4; ++j) {
        int ll = l - 3 + j;
        float v = xz[(size_t)imax_(ll, 0) * 240 + d];   // clamped load
        s += (ll >= 0 ? v : 0.f) * w[d * 4 + j];
    }
    u[idx] = s * sigf_(s);
}

// dt[l,d] = softplus( xdb[l,0:4] . dt_w[d,:] + dt_b[d] )
__global__ void k_dt(const float* __restrict__ xdb, const float* __restrict__ dtw,
                     const float* __restrict__ dtb, float* __restrict__ dt, int L)
{
    int idx = blockIdx.x * blockDim.x + threadIdx.x;
    if (idx >= L * DIN) return;
    int l = idx / DIN, d = idx - l * DIN;
    float s = dtb[d];
#pragma unroll
    for (int r = 0; r < 4; ++r) s += xdb[(size_t)l * 36 + r] * dtw[d * 4 + r];
    dt[idx] = (s > 20.f) ? s : __logf(1.f + __expf(s));
}

// Sequential selective scan: one wave per channel d, lane = state s (0..15).
// h_l = exp(dt*A)*h_{l-1} + dt*B_l*u ; y = sum_s h*C_l + u*D
__global__ void __launch_bounds__(32)
k_ssm_scan(const float* __restrict__ dt, const float* __restrict__ xdb,
           const float* __restrict__ u, const float* __restrict__ A_log,
           const float* __restrict__ Dp, float* __restrict__ y, int L)
{
    const int d = blockIdx.x;
    const int lane = threadIdx.x;
    const int s = lane & 15;
    const bool active = lane < NST;
    const float A  = active ? -__expf(A_log[d * NST + s]) : 0.f;
    const float Dd = Dp[d];
    float h = 0.f;
    for (int l = 0; l < L; ++l) {
        float dtv = dt[(size_t)l * DIN + d];
        float uv  = u[(size_t)l * DIN + d];
        float Bv  = xdb[(size_t)l * 36 + 4 + s];
        float Cv  = xdb[(size_t)l * 36 + 20 + s];
        h = __expf(dtv * A) * h + dtv * Bv * uv;
        float val = active ? h * Cv : 0.f;
        for (int off = 8; off > 0; off >>= 1) val += __shfl_xor(val, off, 32);
        if (lane == 0) y[(size_t)l * DIN + d] = val + uv * Dd;
    }
}

// y[l,d] *= silu(z[l,d]),  z = xz[:,120:240]
__global__ void k_mul_siluz(float* __restrict__ y, const float* __restrict__ xz, int L)
{
    int idx = blockIdx.x * blockDim.x + threadIdx.x;
    if (idx >= L * DIN) return;
    int l = idx / DIN, d = idx - l * DIN;
    float z = xz[(size_t)l * 240 + 120 + d];
    y[idx] *= z * sigf_(z);
}

// ======================= host orchestration ================================
static inline int ceilDiv(int x, int d) { return (x + d - 1) / d; }

extern "C" void kernel_launch(void* const* d_in, const int* in_sizes, int n_in,
                              void* d_out, int out_size, void* d_ws, size_t ws_size,
                              hipStream_t stream)
{
    (void)in_sizes; (void)n_in; (void)out_size; (void)ws_size;
    auto P = [&](int i) { return (const float*)d_in[i]; };
    const float* x1 = P(0);
    const float* x2 = P(1);
    const float* x3 = P(2);
    // param index map (setup_inputs insertion order, offset by 3 inputs):
    // 3..8 conv_f{1,2,3}_{w,b}; 9..26 pyr{1,2,3}_{w,b}{0,1,2};
    // 27..38 al1_s{0,1,2}_{wq,wk,wv,wo}; 39..50 al2_...;
    // 51..58 att{1,2}_{w1,b1,w2,b2}; 59,60 conv_first; 61..64 pe/mb ln;
    // 65 in_proj 66 conv1d_w 67 conv1d_b 68 x_proj 69 dt_w 70 dt_b 71 A_log
    // 72 D 73 out_proj; 74..79 cl{1,2,3}_{w,b}

    const int HW0 = 128 * 128, HW1 = 64 * 64, HW2 = 32 * 32;
    const int L = HW0;
    float* ws = (float*)d_ws;
    size_t off = 0;
    auto AL = [&](size_t n) { float* p = ws + off; off += n; return p; };

    // xfirst survives both phases; the two arenas below alias each other.
    float* xfirst = AL((size_t)EMB * HW0);
    const size_t phaseMark = off;

    // ---- phase 1 arena: features + attention ----
    float* f1 = AL((size_t)EMB * HW0);
    float* f2 = AL((size_t)EMB * HW0);
    float* f3 = AL((size_t)EMB * HW0);
    float* l1_[3] = { AL((size_t)EMB * HW0), AL((size_t)EMB * HW1), AL((size_t)EMB * HW2) };
    float* l2_[3] = { AL((size_t)EMB * HW0), AL((size_t)EMB * HW1), AL((size_t)EMB * HW2) };
    float* l3_[3] = { AL((size_t)EMB * HW0), AL((size_t)EMB * HW1), AL((size_t)EMB * HW2) };
    float* refT = AL((size_t)EMB * HW0);   // HWC transposes for projections
    float* srcT = AL((size_t)EMB * HW0);
    float* Qf  = AL((size_t)EMB * HW0);
    float* Kf  = AL((size_t)EMB * HW0);
    float* Vf  = AL((size_t)EMB * HW0);
    float* Ao  = AL((size_t)EMB * HW0);
    float* a1  = AL((size_t)EMB * HW0);
    float* a3  = AL((size_t)EMB * HW0);
    float* ft1 = AL((size_t)EMB * HW1);
    float* ft2 = AL((size_t)EMB * HW2);
    float* cat = AL((size_t)2 * EMB * HW0);
    float* hb  = AL((size_t)2 * EMB * HW0);
    float* sg  = AL((size_t)EMB * HW0);
    float* cat3 = AL((size_t)3 * EMB * HW0);

    auto conv = [&](const float* in, int Cin, int Hin, int Win,
                    const float* w, const float* b, float* out, int Cout,
                    int stride, int act, float slope) {
        int Ho = (stride == 1) ? Hin : Hin / 2;
        int Wo = (stride == 1) ? Win : Win / 2;
        dim3 g(ceilDiv(Ho * Wo, 16 * MT_C), ceilDiv(Cout, 16));
        k_conv3x3_wmma<<<g, 32, 0, stream>>>(in, w, b, out, Cin, Hin, Win, Cout,
                                             stride, act, slope);
    };
    auto gemm = [&](const float* A, const float* W, const float* bias, const float* Res,
                    float* Out, int M, int N, int K, int sAm, int sCm, int sCn) {
        dim3 g(ceilDiv(M, 16 * MT_G), ceilDiv(N, 16));
        k_gemm_wmma<<<g, 32, 0, stream>>>(A, W, bias, Res, Out, M, N, K, sAm, sCm, sCn);
    };

    // features + pyramids
    conv(x1, 8, 128, 128, P(3), P(4), f1, EMB, 1, 0, 0.f);
    conv(x2, 8, 128, 128, P(5), P(6), f2, EMB, 1, 0, 0.f);
    conv(x3, 8, 128, 128, P(7), P(8), f3, EMB, 1, 0, 0.f);
    auto pyr = [&](const float* f, int base, float* const o[3]) {
        conv(f,    EMB, 128, 128, P(base + 0), P(base + 1), o[0], EMB, 1, 1, 0.1f);
        conv(o[0], EMB, 128, 128, P(base + 2), P(base + 3), o[1], EMB, 2, 1, 0.1f);
        conv(o[1], EMB, 64,  64,  P(base + 4), P(base + 5), o[2], EMB, 2, 1, 0.1f);
    };
    pyr(f1, 9, l1_);
    pyr(f2, 15, l2_);
    pyr(f3, 21, l3_);

    // _align(ref=l2, src=lX): coarse->fine with up2 accumulate
    auto align = [&](float* const ref[3], float* const src[3], int base, float* outFull) {
        const int Hs[3] = { 128, 64, 32 };
        float* featPrev = nullptr;
        for (int s = 2; s >= 0; --s) {
            int Hh = Hs[s], HW = Hh * Hh;
            const float* wq = P(base + s * 4 + 0);
            const float* wk = P(base + s * 4 + 1);
            const float* wv = P(base + s * 4 + 2);
            const float* wo = P(base + s * 4 + 3);
            // transpose CHW -> HWC so projections use the contiguous-K path
            int nT = EMB * HW;
            k_chw2hwc<<<(nT + 255) / 256, 256, 0, stream>>>(ref[s], refT, EMB, HW);
            k_chw2hwc<<<(nT + 255) / 256, 256, 0, stream>>>(src[s], srcT, EMB, HW);
            gemm(refT, wq, nullptr, nullptr, Qf, HW, EMB, EMB, EMB, EMB, 1);
            gemm(srcT, wk, nullptr, nullptr, Kf, HW, EMB, EMB, EMB, EMB, 1);
            gemm(srcT, wv, nullptr, nullptr, Vf, HW, EMB, EMB, EMB, EMB, 1);
            int nwin = (Hh / 8) * (Hh / 8);
            k_win_attn<<<nwin, 256, 0, stream>>>(Qf, Kf, Vf, Ao, Hh, Hh);
            float* dst = (s == 0) ? outFull : ((s == 1) ? ft1 : ft2);
            // O = Ao @ wo^T  -> CHW, fused +src residual
            gemm(Ao, wo, nullptr, src[s], dst, HW, EMB, EMB, EMB, 1, HW);
            if (featPrev) {
                int n = EMB * HW;
                k_add_up2<<<(n + 255) / 256, 256, 0, stream>>>(dst, featPrev, EMB, Hh, Hh);
            }
            featPrev = dst;
        }
    };
    align(l2_, l1_, 27, a1);
    align(l2_, l3_, 39, a3);

    // f1a = f2 + a1 ; f3a = f2 + a3 (in place over a1/a3)
    k_add<<<(EMB * HW0 + 255) / 256, 256, 0, stream>>>(a1, f2, EMB * HW0);
    k_add<<<(EMB * HW0 + 255) / 256, 256, 0, stream>>>(a3, f2, EMB * HW0);

    auto spatialAtt = [&](float* xa, const float* xb, int base) {
        hipMemcpyAsync(cat, xa, (size_t)EMB * HW0 * sizeof(float),
                       hipMemcpyDeviceToDevice, stream);
        hipMemcpyAsync(cat + (size_t)EMB * HW0, xb, (size_t)EMB * HW0 * sizeof(float),
                       hipMemcpyDeviceToDevice, stream);
        conv(cat, 2 * EMB, 128, 128, P(base + 0), P(base + 1), hb, 2 * EMB, 1, 1, 0.01f);
        conv(hb,  2 * EMB, 128, 128, P(base + 2), P(base + 3), sg, EMB, 1, 2, 0.f);
        k_mul<<<(EMB * HW0 + 255) / 256, 256, 0, stream>>>(xa, sg, EMB * HW0);
    };
    spatialAtt(a1, f2, 51);
    spatialAtt(a3, f2, 55);

    // conv_first over concat[f1a, f2, f3a] (180 ch)
    hipMemcpyAsync(cat3, a1, (size_t)EMB * HW0 * sizeof(float), hipMemcpyDeviceToDevice, stream);
    hipMemcpyAsync(cat3 + (size_t)EMB * HW0, f2, (size_t)EMB * HW0 * sizeof(float),
                   hipMemcpyDeviceToDevice, stream);
    hipMemcpyAsync(cat3 + (size_t)2 * EMB * HW0, a3, (size_t)EMB * HW0 * sizeof(float),
                   hipMemcpyDeviceToDevice, stream);
    conv(cat3, 3 * EMB, 128, 128, P(59), P(60), xfirst, EMB, 1, 0, 0.f);

    // ---- phase 2 arena (aliases phase-1 buffers; stream order serializes) ----
    off = phaseMark;
    float* tok = AL((size_t)L * EMB);
    float* xz  = AL((size_t)L * 240);
    float* ub  = AL((size_t)L * DIN);
    float* xdb = AL((size_t)L * 36);
    float* dtb = AL((size_t)L * DIN);
    float* yb  = AL((size_t)L * DIN);
    float* x2b = AL((size_t)EMB * HW0);
    float* c1  = AL((size_t)30 * HW0);
    float* c2  = AL((size_t)15 * HW0);

    // tokens: pe_ln then mb_ln (residual in reference is dead code)
    k_double_ln<<<(L + 255) / 256, 256, 0, stream>>>(xfirst, P(61), P(62), P(63), P(64), tok, L);
    // xz = tok @ in_proj^T : [L,240]
    gemm(tok, P(65), nullptr, nullptr, xz, L, 240, EMB, EMB, 240, 1);
    // depthwise causal conv + silu -> u [L,120]
    k_conv1d_silu<<<(L * DIN + 255) / 256, 256, 0, stream>>>(xz, P(66), P(67), ub, L);
    // xdb = u @ x_proj^T : [L,36]
    gemm(ub, P(68), nullptr, nullptr, xdb, L, 36, DIN, DIN, 36, 1);
    // dt = softplus(xdb[:, :4] @ dt_w^T + dt_b)
    k_dt<<<(L * DIN + 255) / 256, 256, 0, stream>>>(xdb, P(69), P(70), dtb, L);
    // selective scan
    k_ssm_scan<<<DIN, 32, 0, stream>>>(dtb, xdb, ub, P(71), P(72), yb, L);
    // y *= silu(z)
    k_mul_siluz<<<(L * DIN + 255) / 256, 256, 0, stream>>>(yb, xz, L);
    // x = (y) @ out_proj^T  -> CHW
    gemm(yb, P(73), nullptr, nullptr, x2b, L, EMB, DIN, DIN, 1, HW0);

    // tail convs
    conv(x2b, EMB, 128, 128, P(74), P(75), c1, 30, 1, 1, 0.2f);
    conv(c1,  30,  128, 128, P(76), P(77), c2, 15, 1, 1, 0.2f);
    conv(c2,  15,  128, 128, P(78), P(79), (float*)d_out, 4, 1, 0, 0.f);
}
